// MultiHeadSelfAttention_15229954031764
// MI455X (gfx1250) — matrix operand order repaired and checked by reading, not compiled, not run on hardware
//
#include <hip/hip_runtime.h>
#include <hip/hip_bf16.h>

// ---------------------------------------------------------------------------
// MultiHeadSelfAttention on MI455X (gfx1250).
//   bf16 WMMA everywhere, flash softmax, async global->LDS tile staging
//   (GLOBAL_LOAD_ASYNC_TO_LDS_B128 + s_wait_asynccnt, double buffered).
// B=4, S=2048, D=1024, H=16, Hd=64.
// ---------------------------------------------------------------------------

typedef __attribute__((ext_vector_type(16))) __bf16         v16bf;
typedef __attribute__((ext_vector_type(8)))  float          v8f;
typedef __attribute__((ext_vector_type(8)))  unsigned short v8u;
typedef unsigned short u16;

#define BB   4
#define SS   2048
#define DD   1024
#define HH   16
#define HD   64
#define MM   (BB * SS)          // 8192 rows

union FragU { v16bf bf; v8u u[2]; };

__device__ __forceinline__ u16 f2bf(float f) {
    unsigned int u = __float_as_uint(f);
    unsigned int r = u + 0x7FFFu + ((u >> 16) & 1u);   // round-to-nearest-even
    return (u16)(r >> 16);
}

__device__ __forceinline__ v8f wmma_bf16(v16bf a, v16bf b, v8f c) {
    return __builtin_amdgcn_wmma_f32_16x16x32_bf16(
        false, a, false, b, (short)0, c, false, false);
}

// Async DMA one 16-byte chunk global -> LDS (ASYNCcnt-tracked).
__device__ __forceinline__ void async_g2l_b128(u16* lds_dst, const u16* gsrc) {
    unsigned loff = (unsigned)(size_t)lds_dst;   // low 32 bits = LDS byte offset
    asm volatile("global_load_async_to_lds_b128 %0, %1, off"
                 :: "v"(loff), "v"(gsrc) : "memory");
}
#define WAIT_ASYNC0()  asm volatile("s_wait_asynccnt 0x0" ::: "memory")
#define WAIT_DS0()     asm volatile("s_wait_dscnt 0x0"    ::: "memory")

// A-frag: 16x32 tile, base = &mat[row0*ld + k0], row-major, K contiguous.
__device__ __forceinline__ v16bf load_frag_a(const u16* base, int ld, int lane) {
    int m = lane & 15, hf = lane >> 4;
    const u16* p = base + (size_t)m * ld + hf * 16;
    FragU f;
    f.u[0] = *(const v8u*)(p);
    f.u[1] = *(const v8u*)(p + 8);
    return f.bf;
}

// B-frag from a K-contiguous row (global or LDS): p = &row[n][k0 + hf*16].
__device__ __forceinline__ v16bf frag_b(const u16* p) {
    FragU f;
    f.u[0] = *(const v8u*)(p);
    f.u[1] = *(const v8u*)(p + 8);
    return f.bf;
}

// ---------------------------------------------------------------------------
// fp32 -> bf16 cast (vectorized x4)
// ---------------------------------------------------------------------------
__global__ void cast_bf16_kernel(const float* __restrict__ in, u16* __restrict__ out, int n4) {
    int i = blockIdx.x * blockDim.x + threadIdx.x;
    if (i >= n4) return;
    const float4 v = ((const float4*)in)[i];
    ushort4 o;
    o.x = f2bf(v.x); o.y = f2bf(v.y); o.z = f2bf(v.z); o.w = f2bf(v.w);
    ((ushort4*)out)[i] = o;
}

// ---------------------------------------------------------------------------
// Projection GEMM body:  block = 8 waves, each wave a 32x64 tile of X @ W^T.
// All waves share n0, so W chunks (64 rows x 128 k, 16 KB) are staged in LDS
// with async DMA, double buffered.  X A-frags stream from global.
// Grid: 32 m-groups (256 rows) x 16 n-tiles = 512 blocks x 256 threads.
// ---------------------------------------------------------------------------
#define PROJ_BODY(ACC)                                                          \
    __shared__ __align__(16) u16 wbuf[2][64 * 128];                             \
    int tid = threadIdx.x;                                                      \
    int lane = tid & 31, wslot = tid >> 5;                                      \
    int m_group = blockIdx.x >> 4;                                              \
    int n_tile  = blockIdx.x & 15;                                              \
    int m0 = m_group * 256 + wslot * 32;                                        \
    int n0 = n_tile << 6;                                                       \
    int l16 = lane & 15, hf = lane >> 4;                                        \
    v8f ACC[2][4] = {{v8f{}, v8f{}, v8f{}, v8f{}},                              \
                     {v8f{}, v8f{}, v8f{}, v8f{}}};                             \
    /* stage 0 */                                                               \
    _Pragma("unroll")                                                           \
    for (int it = 0; it < 4; ++it) {                                            \
        int i = it * 256 + tid;                                                 \
        int row = i >> 4, col = (i & 15) * 8;                                   \
        async_g2l_b128(&wbuf[0][row * 128 + col],                               \
                       W + (size_t)(n0 + row) * DD + col);                      \
    }                                                                           \
    for (int kc = 0; kc < DD; kc += 128) {                                      \
        int cur = (kc >> 7) & 1;                                                \
        WAIT_ASYNC0();                                                          \
        __syncthreads();                                                        \
        if (kc + 128 < DD) {                                                    \
            _Pragma("unroll")                                                   \
            for (int it = 0; it < 4; ++it) {                                    \
                int i = it * 256 + tid;                                         \
                int row = i >> 4, col = (i & 15) * 8;                           \
                async_g2l_b128(&wbuf[cur ^ 1][row * 128 + col],                 \
                               W + (size_t)(n0 + row) * DD + kc + 128 + col);   \
            }                                                                   \
        }                                                                       \
        __builtin_prefetch(X + (size_t)m0 * DD + kc + 128, 0, 0);               \
        _Pragma("unroll")                                                       \
        for (int kk = 0; kk < 128; kk += 32) {                                  \
            int k0 = kc + kk;                                                   \
            v16bf a0 = load_frag_a(X + (size_t)(m0     ) * DD + k0, DD, lane);  \
            v16bf a1 = load_frag_a(X + (size_t)(m0 + 16) * DD + k0, DD, lane);  \
            const u16* wp = &wbuf[cur][kk];                                     \
            v16bf b0 = frag_b(wp + (     l16) * 128 + hf * 16);                 \
            v16bf b1 = frag_b(wp + (16 + l16) * 128 + hf * 16);                 \
            v16bf b2 = frag_b(wp + (32 + l16) * 128 + hf * 16);                 \
            v16bf b3 = frag_b(wp + (48 + l16) * 128 + hf * 16);                 \
            ACC[0][0] = wmma_bf16(a0, b0, ACC[0][0]);                           \
            ACC[0][1] = wmma_bf16(a0, b1, ACC[0][1]);                           \
            ACC[0][2] = wmma_bf16(a0, b2, ACC[0][2]);                           \
            ACC[0][3] = wmma_bf16(a0, b3, ACC[0][3]);                           \
            ACC[1][0] = wmma_bf16(a1, b0, ACC[1][0]);                           \
            ACC[1][1] = wmma_bf16(a1, b1, ACC[1][1]);                           \
            ACC[1][2] = wmma_bf16(a1, b2, ACC[1][2]);                           \
            ACC[1][3] = wmma_bf16(a1, b3, ACC[1][3]);                           \
        }                                                                       \
    }

// bf16-out projection: vmode 0 -> Q/K [B,H,S,64]; vmode 1 -> Vt [B,H,64,S]
__global__ __launch_bounds__(256)
void proj_bf16_kernel(const u16* __restrict__ X, const u16* __restrict__ W,
                      const float* __restrict__ bias, u16* __restrict__ out, int vmode) {
    PROJ_BODY(acc)
    int b_idx = m0 >> 11;
    int s_base = m0 & (SS - 1);
    int head = n0 >> 6;
#pragma unroll
    for (int i = 0; i < 2; ++i) {
#pragma unroll
        for (int t = 0; t < 4; ++t) {
            int d = t * 16 + l16;
            float bia = bias[n0 + t * 16 + l16];
#pragma unroll
            for (int r = 0; r < 8; ++r) {
                int s = s_base + i * 16 + r + 8 * hf;
                float v = acc[i][t][r] + bia;
                size_t idx;
                if (vmode == 0) idx = ((size_t)(b_idx * HH + head) * SS + s) * HD + d;
                else            idx = ((size_t)(b_idx * HH + head) * HD + d) * SS + s;
                out[idx] = f2bf(v);
            }
        }
    }
}

// fp32-out projection: out = O @ wo^T + bo, [MM, DD] row-major
__global__ __launch_bounds__(256)
void proj_f32_kernel(const u16* __restrict__ X, const u16* __restrict__ W,
                     const float* __restrict__ bias, float* __restrict__ out) {
    PROJ_BODY(acc)
#pragma unroll
    for (int i = 0; i < 2; ++i) {
#pragma unroll
        for (int t = 0; t < 4; ++t) {
            float bia = bias[n0 + t * 16 + l16];
#pragma unroll
            for (int r = 0; r < 8; ++r) {
                int m = m0 + i * 16 + r + 8 * hf;
                out[(size_t)m * DD + n0 + t * 16 + l16] = acc[i][t][r] + bia;
            }
        }
    }
}

// ---------------------------------------------------------------------------
// Flash attention.  Block = 8 waves, all sharing one (b,h); each wave owns a
// 16-query tile.  K (32x64) and V (64x32) tiles for the current 32-key block
// are staged in LDS by async DMA (double buffered) and shared by all 8 waves.
// ---------------------------------------------------------------------------
__global__ __launch_bounds__(256)
void flash_attn_kernel(const u16* __restrict__ Q, const u16* __restrict__ K,
                       const u16* __restrict__ Vt, u16* __restrict__ O) {
    __shared__ __align__(16) u16 kbuf[2][32 * 64];   // 2 x 4 KB
    __shared__ __align__(16) u16 vbuf[2][64 * 32];   // 2 x 4 KB
    __shared__ __align__(16) u16 pbuf[8][16 * 32];   // 8 KB (per-wave P tiles)

    int tid   = threadIdx.x;
    int lane  = tid & 31;
    int wslot = tid >> 5;
    int wave  = blockIdx.x * 8 + wslot;
    int qt = wave & 127;           // blocks never straddle (b,h): 8 | 128
    int bh = wave >> 7;
    int q0 = qt << 4;
    int l16 = lane & 15, hf = lane >> 4;

    const u16* Qb = Q  + (size_t)bh * SS * HD;
    const u16* Kb = K  + (size_t)bh * SS * HD;
    const u16* Vb = Vt + (size_t)bh * HD * SS;
    u16* wls = pbuf[wslot];

    // cooperative async stage of one 32-key tile (1 K-chunk + 1 V-chunk/thread)
    int krow = tid >> 3, kcol = (tid & 7) * 8;   // K: 32 rows x 64, 8 chunks/row
    int vrow = tid >> 2, vcol = (tid & 3) * 8;   // V: 64 rows x 32, 4 chunks/row
#define ISSUE_TILE(buf, kb)                                                        \
    do {                                                                           \
        async_g2l_b128(&kbuf[buf][krow * 64 + kcol],                               \
                       Kb + (size_t)((kb) + krow) * HD + kcol);                    \
        async_g2l_b128(&vbuf[buf][vrow * 32 + vcol],                               \
                       Vb + (size_t)vrow * SS + (kb) + vcol);                      \
    } while (0)

    v16bf qa0 = load_frag_a(Qb + (size_t)q0 * HD +  0, HD, lane);
    v16bf qa1 = load_frag_a(Qb + (size_t)q0 * HD + 32, HD, lane);

    float mrun[8], lrun[8];
#pragma unroll
    for (int r = 0; r < 8; ++r) { mrun[r] = -1e30f; lrun[r] = 0.0f; }
    v8f oacc[4] = {v8f{}, v8f{}, v8f{}, v8f{}};

    ISSUE_TILE(0, 0);
    for (int kb = 0; kb < SS; kb += 32) {
        int cur = (kb >> 5) & 1;
        WAIT_ASYNC0();                         // tile kb landed (issued last iter)
        __syncthreads();                       // visible to all; prior reads done
        if (kb + 32 < SS) ISSUE_TILE(cur ^ 1, kb + 32);  // overlap next fetch

        // ---- scores: 16x32 tile via 4 WMMA, K-frags from LDS
        const u16* kp = kbuf[cur];
        v16bf k0a = frag_b(kp + (     l16) * 64 +      hf * 16);
        v16bf k0b = frag_b(kp + (     l16) * 64 + 32 + hf * 16);
        v16bf k1a = frag_b(kp + (16 + l16) * 64 +      hf * 16);
        v16bf k1b = frag_b(kp + (16 + l16) * 64 + 32 + hf * 16);
        v8f s0 = {}, s1 = {};
        s0 = wmma_bf16(qa0, k0a, s0);
        s0 = wmma_bf16(qa1, k0b, s0);
        s1 = wmma_bf16(qa0, k1a, s1);
        s1 = wmma_bf16(qa1, k1b, s1);

        // ---- V frags issued early (ds loads overlap softmax VALU work)
        const u16* vp = vbuf[cur];
        v16bf vb0 = frag_b(vp + (     l16) * 32 + hf * 16);
        v16bf vb1 = frag_b(vp + (16 + l16) * 32 + hf * 16);
        v16bf vb2 = frag_b(vp + (32 + l16) * 32 + hf * 16);
        v16bf vb3 = frag_b(vp + (48 + l16) * 32 + hf * 16);

        // ---- online softmax (rows r+8*hf, stats replicated over 16 lanes)
#pragma unroll
        for (int r = 0; r < 8; ++r) {
            float a = s0[r] * 0.125f;            // 1/sqrt(64)
            float b = s1[r] * 0.125f;
            float mx = fmaxf(a, b);
            mx = fmaxf(mx, __shfl_xor(mx, 1, 32));
            mx = fmaxf(mx, __shfl_xor(mx, 2, 32));
            mx = fmaxf(mx, __shfl_xor(mx, 4, 32));
            mx = fmaxf(mx, __shfl_xor(mx, 8, 32));
            float mnew  = fmaxf(mrun[r], mx);
            float alpha = __expf(mrun[r] - mnew);
            float p0 = __expf(a - mnew);
            float p1 = __expf(b - mnew);
            float rs = p0 + p1;
            rs += __shfl_xor(rs, 1, 32);
            rs += __shfl_xor(rs, 2, 32);
            rs += __shfl_xor(rs, 4, 32);
            rs += __shfl_xor(rs, 8, 32);
            lrun[r] = lrun[r] * alpha + rs;
            mrun[r] = mnew;
            oacc[0][r] *= alpha; oacc[1][r] *= alpha;
            oacc[2][r] *= alpha; oacc[3][r] *= alpha;
            int row = r + 8 * hf;
            wls[row * 32 + l16]      = f2bf(p0);
            wls[row * 32 + 16 + l16] = f2bf(p1);
        }
        WAIT_DS0();                              // P tile RAW within wave

        // ---- O += P @ V
        FragU pf;
        pf.u[0] = *(const v8u*)(&wls[l16 * 32 + hf * 16]);
        pf.u[1] = *(const v8u*)(&wls[l16 * 32 + hf * 16 + 8]);
        v16bf pa = pf.bf;
        oacc[0] = wmma_bf16(pa, vb0, oacc[0]);
        oacc[1] = wmma_bf16(pa, vb1, oacc[1]);
        oacc[2] = wmma_bf16(pa, vb2, oacc[2]);
        oacc[3] = wmma_bf16(pa, vb3, oacc[3]);
    }
#undef ISSUE_TILE

    // ---- epilogue: normalize, write O[b, s, h*64+d] bf16
    int b_idx = bh >> 4, head = bh & 15;
#pragma unroll
    for (int t = 0; t < 4; ++t) {
#pragma unroll
        for (int r = 0; r < 8; ++r) {
            int s   = q0 + r + 8 * hf;
            int col = head * HD + t * 16 + l16;
            float v = oacc[t][r] / lrun[r];
            O[((size_t)b_idx * SS + s) * DD + col] = f2bf(v);
        }
    }
}

// ---------------------------------------------------------------------------
extern "C" void kernel_launch(void* const* d_in, const int* in_sizes, int n_in,
                              void* d_out, int out_size, void* d_ws, size_t ws_size,
                              hipStream_t stream) {
    const float* x  = (const float*)d_in[0];
    const float* wq = (const float*)d_in[1];
    const float* bq = (const float*)d_in[2];
    const float* wk = (const float*)d_in[3];
    const float* bk = (const float*)d_in[4];
    const float* wv = (const float*)d_in[5];
    const float* bv = (const float*)d_in[6];
    const float* wo = (const float*)d_in[7];
    const float* bo = (const float*)d_in[8];

    // Workspace layout (bytes). ~72 MB total; O reuses the xb region.
    char* ws = (char*)d_ws;
    const size_t XB   = (size_t)MM * DD * 2;   // 16 MB  bf16 x  (later reused as O)
    const size_t WB   = (size_t)DD * DD * 2;   //  2 MB per weight
    const size_t QKVB = (size_t)MM * DD * 2;   // 16 MB per Q/K/Vt
    u16* xb  = (u16*)(ws);
    u16* wqb = (u16*)(ws + XB);
    u16* wkb = (u16*)(ws + XB + WB);
    u16* wvb = (u16*)(ws + XB + 2 * WB);
    u16* wob = (u16*)(ws + XB + 3 * WB);
    u16* Qb  = (u16*)(ws + XB + 4 * WB);
    u16* Kb  = (u16*)(ws + XB + 4 * WB + QKVB);
    u16* Vtb = (u16*)(ws + XB + 4 * WB + 2 * QKVB);
    u16* Ob  = xb;                              // x no longer needed after QKV

    // 1) fp32 -> bf16 casts
    {
        int n4 = (MM * DD) / 4;
        cast_bf16_kernel<<<(n4 + 255) / 256, 256, 0, stream>>>(x, xb, n4);
        int w4 = (DD * DD) / 4;
        cast_bf16_kernel<<<(w4 + 255) / 256, 256, 0, stream>>>(wq, wqb, w4);
        cast_bf16_kernel<<<(w4 + 255) / 256, 256, 0, stream>>>(wk, wkb, w4);
        cast_bf16_kernel<<<(w4 + 255) / 256, 256, 0, stream>>>(wv, wvb, w4);
        cast_bf16_kernel<<<(w4 + 255) / 256, 256, 0, stream>>>(wo, wob, w4);
    }

    // 2) QKV projections: 512 blocks x 256 threads (8 waves, W staged in LDS)
    proj_bf16_kernel<<<512, 256, 0, stream>>>(xb, wqb, bq, Qb, 0);
    proj_bf16_kernel<<<512, 256, 0, stream>>>(xb, wkb, bk, Kb, 0);
    proj_bf16_kernel<<<512, 256, 0, stream>>>(xb, wvb, bv, Vtb, 1);

    // 3) Flash attention: 8192 q-tiles -> 1024 blocks x 8 waves
    flash_attn_kernel<<<1024, 256, 0, stream>>>(Qb, Kb, Vtb, Ob);

    // 4) Output projection to fp32 d_out
    proj_f32_kernel<<<512, 256, 0, stream>>>(Ob, wob, bo, (float*)d_out);
}
